// DecoderV2_75694503624764
// MI455X (gfx1250) — compile-verified, matrix operand order checked by reading, not run on hardware
//
#include <hip/hip_runtime.h>
#include <hip/hip_bf16.h>

// ---------------------------------------------------------------------------
// Problem constants (from reference)
// ---------------------------------------------------------------------------
#define BATCH   1024
#define TPRED   24
#define NFILT   64
#define HID     128
#define NDIL    6
#define LENC    168
#define NFEAT   15

// fragment pool layout (each fragment = 32 lanes x 16 bf16 = 512 u16)
#define FRAG_W1    0          // 4 frags  : W1 padded to K=32, N=64
#define FRAG_WCAT  4          // 32 frags : [W2;W3] K=128, N=128
#define FRAG_W4    36         // 16 frags : W4 K=64, N=128
#define FRAG_W5    52         // 96 frags : W5 K=384, N=128
#define NFRAG_TOT  148
#define APP_OFF    (160*1024) // appended bf16 [NDIL][B][T][64] after frag pool

// LDS strides (padded to break 64-bank stride conflicts)
#define LDA 132   // 16 x 128 activation tile [state|inputs]
#define LDG 68    // 16 x 64 gated / relu-skip tile

typedef __attribute__((ext_vector_type(16))) __bf16        bf16x16;
typedef __attribute__((ext_vector_type(8)))  float         f32x8;
typedef __attribute__((ext_vector_type(8)))  unsigned int  u32x8;

#define LOG2E  1.4426950408889634f

// hardware float->bf16 conversion (single v_cvt op on gfx1250)
__device__ __forceinline__ unsigned short f2bf(float f) {
    return __builtin_bit_cast(unsigned short, (__bf16)f);
}

// branch-free fast transcendentals on the TRANS pipe (v_exp_f32 / v_rcp_f32)
__device__ __forceinline__ float fast_sigmoid(float x) {
    return __builtin_amdgcn_rcpf(1.f + __builtin_amdgcn_exp2f(-LOG2E * x));
}
__device__ __forceinline__ float fast_tanh(float x) {
    // tanh(x) = 1 - 2/(exp(2x)+1); saturates correctly via inf/rcp
    return 1.f - 2.f * __builtin_amdgcn_rcpf(
                     1.f + __builtin_amdgcn_exp2f(2.f * LOG2E * x));
}

__device__ __forceinline__ f32x8 wmma_bf16(bf16x16 a, bf16x16 b, f32x8 c) {
    return __builtin_amdgcn_wmma_f32_16x16x32_bf16(
        false, a, false, b, (short)0, c, false, false);
}

// load one pre-swizzled fragment: 32 bytes per lane, contiguous
__device__ __forceinline__ bf16x16 ldfrag16(const unsigned short* p) {
    const uint4* q = (const uint4*)p;
    uint4 a = q[0], b = q[1];
    u32x8 u;
    u[0]=a.x; u[1]=a.y; u[2]=a.z; u[3]=a.w;
    u[4]=b.x; u[5]=b.y; u[6]=b.z; u[7]=b.w;
    return __builtin_bit_cast(bf16x16, u);
}

// build an A-fragment (16xK tile, K-offset ktOff) from a row-major bf16 LDS
// buffer with leading dim `ld`.  ISA layout: lane m=l&15, halves split K.
__device__ __forceinline__ bf16x16 loadA(const unsigned short* base, int ld,
                                         int m, int hk, int ktOff) {
    const unsigned short* row = base + m*ld + ktOff + hk*8;
    u32x8 u;
#pragma unroll
    for (int v = 0; v < 8; ++v) {
        int kb = 2*v + ((v >= 4) ? 8 : 0);      // K pair base within half
        u[v] = *(const unsigned int*)(row + kb);
    }
    return __builtin_bit_cast(bf16x16, u);
}

// ---------------------------------------------------------------------------
// Prep: pack all weights into bf16 WMMA fragments in workspace
// ---------------------------------------------------------------------------
__global__ void pack_weights(const float* __restrict__ W1,
                             const float* __restrict__ W2,
                             const float* __restrict__ W3,
                             const float* __restrict__ W4,
                             const float* __restrict__ W5,
                             unsigned short* __restrict__ frag) {
    int g = blockIdx.x * blockDim.x + threadIdx.x;
    if (g >= NFRAG_TOT * 512) return;
    int f    = g >> 9;
    int lane = (g >> 4) & 31;
    int s    = g & 15;
    // K index inside a 32-wide tile for (lane, slot) per 16-bit A/B layout
    int kin = 2*(s >> 1) + ((s >= 8) ? 8 : 0) + ((lane & 16) ? 8 : 0) + (s & 1);
    int n15 = lane & 15;
    float val = 0.f;
    if (f < FRAG_WCAT) {                       // W1, K padded 16->32
        int nt = f - FRAG_W1;
        int k = kin, n = nt*16 + n15;
        if (k < 16) val = W1[k*NFILT + n];
    } else if (f < FRAG_W4) {                  // [W2 ; W3]
        int fl = f - FRAG_WCAT, kt = fl >> 3, nt = fl & 7;
        int k = kt*32 + kin, n = nt*16 + n15;
        val = (k < 64) ? W2[k*(2*NFILT) + n] : W3[(k-64)*(2*NFILT) + n];
    } else if (f < FRAG_W5) {                  // W4
        int fl = f - FRAG_W4, kt = fl >> 3, nt = fl & 7;
        int k = kt*32 + kin, n = nt*16 + n15;
        val = W4[k*(2*NFILT) + n];
    } else {                                   // W5
        int fl = f - FRAG_W5, kt = fl >> 3, nt = fl & 7;
        int k = kt*32 + kin, n = nt*16 + n15;
        val = W5[k*HID + n];
    }
    frag[g] = f2bf(val);
}

// ---------------------------------------------------------------------------
// Main decoder: 16 blocks x 128 threads (4 waves); wave = 16 batch rows
// ---------------------------------------------------------------------------
__global__ __launch_bounds__(128, 1)
void decode_kernel(const float* __restrict__ feat,
                   const float* __restrict__ init,
                   const float* __restrict__ enc,
                   const float* __restrict__ b1, const float* __restrict__ b2,
                   const float* __restrict__ b4, const float* __restrict__ b5,
                   const float* __restrict__ W6, const float* __restrict__ b6,
                   const unsigned short* __restrict__ frag,
                   unsigned short* __restrict__ app,
                   float* __restrict__ out) {
    __shared__ __align__(16) unsigned short lWcat[32 * 512];  // 32 KB
    __shared__ __align__(16) unsigned short sA[4][16 * LDA];  // [state|inputs]
    __shared__ __align__(16) unsigned short sG[4][16 * LDG];  // gated / skips
    __shared__ float sY[4][16];

    const int tid  = threadIdx.x;
    const int wave = tid >> 5, lane = tid & 31;
    const int m = lane & 15, hk = lane >> 4;
    const int r0 = (blockIdx.x * 4 + wave) * 16;   // first batch row of tile

    // stage the hot [W2;W3] fragment block into LDS (shared by all waves)
    {
        const uint4* src = (const uint4*)(frag + FRAG_WCAT * 512);
        uint4*       dst = (uint4*)lWcat;
        for (int i = tid; i < (32 * 512) / 8; i += 128) dst[i] = src[i];
    }
    __syncthreads();

    if (lane < 16) sY[wave][lane] = init[r0 + lane];
    const float b6v = b6[0];

    // ---- hoist all t-invariant per-lane bias / W6 scalars into registers ----
    float b1v[4], b2f[4], b2g[4], b4s[4], b4r[4], b5v[8], w6v[8];
#pragma unroll
    for (int j = 0; j < 4; ++j) {
        b1v[j] = b1[j*16 + m];
        b2f[j] = b2[j*16 + m];
        b2g[j] = b2[64 + j*16 + m];
        b4s[j] = b4[j*16 + m];
        b4r[j] = b4[64 + j*16 + m];
    }
#pragma unroll
    for (int nt = 0; nt < 8; ++nt) {
        b5v[nt] = b5[nt*16 + m];
        w6v[nt] = W6[nt*16 + m];
    }

#pragma unroll 1
    for (int t = 0; t < TPRED; ++t) {
        asm volatile("s_wait_dscnt 0" ::: "memory");

        // -------- inputs = tanh([prev, feat_t] @ W1 + b1) --------
        f32x8 acc_in[4];
        {
            const float* fr = feat + (size_t)(r0 + m) * TPRED * NFEAT + t * NFEAT;
            u32x8 ua;
#pragma unroll
            for (int v = 0; v < 8; ++v) {
                int kb = 2*v + ((v >= 4) ? 8 : 0) + hk*8;
                float x0 = 0.f, x1 = 0.f;
                if (kb < 16) {                     // cur[k]: k0 = prev, rest = feat
                    x0 = (kb == 0) ? sY[wave][m] : fr[kb - 1];
                    x1 = fr[kb];
                }
                ua[v] = (unsigned int)f2bf(x0) | ((unsigned int)f2bf(x1) << 16);
            }
            bf16x16 a = __builtin_bit_cast(bf16x16, ua);
#pragma unroll
            for (int nt = 0; nt < 4; ++nt) {
                f32x8 c = {};
                bf16x16 b = ldfrag16(frag + (size_t)(FRAG_W1 + nt) * 512 + lane * 16);
                c = wmma_bf16(a, b, c);
#pragma unroll
                for (int v = 0; v < 8; ++v) c[v] = fast_tanh(c[v] + b1v[nt]);
                acc_in[nt] = c;
            }
        }
        // write inputs into sA cols 64..127 (A-operand staging)
#pragma unroll
        for (int nt = 0; nt < 4; ++nt)
#pragma unroll
            for (int v = 0; v < 8; ++v)
                sA[wave][(v + hk*8)*LDA + 64 + nt*16 + m] = f2bf(acc_in[nt][v]);

        f32x8 acc_h[8];
#pragma unroll
        for (int nt = 0; nt < 8; ++nt) acc_h[nt] = (f32x8){};

        // ---------------- dilation stack ----------------
#pragma unroll 1
        for (int i = 0; i < NDIL; ++i) {
            const int d = 1 << i;
            // state -> sA cols 0..63
            if (t >= d) {
                const unsigned short* ap =
                    app + ((size_t)i * BATCH + r0) * TPRED * 64;
                for (int e = lane; e < 1024; e += 32) {
                    int rr = e >> 6, c = e & 63;
                    sA[wave][rr*LDA + c] =
                        ap[((size_t)rr * TPRED + (t - d)) * 64 + c];
                }
            } else {
                int idx = LENC + t - d;            // t<d  =>  in [136,167]
                const float* ep =
                    enc + (((size_t)i * BATCH + r0) * LENC + idx) * 64;
                for (int e = lane; e < 1024; e += 32) {
                    int rr = e >> 6, c = e & 63;
                    sA[wave][rr*LDA + c] = f2bf(ep[(size_t)rr * LENC * 64 + c]);
                }
            }
            asm volatile("s_wait_dscnt 0" ::: "memory");

            // dilated = [state|inputs] @ [W2;W3]   (A reused across 8 N-tiles)
            f32x8 accd[8];
#pragma unroll
            for (int nt = 0; nt < 8; ++nt) accd[nt] = (f32x8){};
#pragma unroll
            for (int kt = 0; kt < 4; ++kt) {
                bf16x16 a = loadA(&sA[wave][0], LDA, m, hk, kt*32);
#pragma unroll
                for (int nt = 0; nt < 8; ++nt) {
                    bf16x16 b = ldfrag16(lWcat + (kt*8 + nt)*512 + lane*16);
                    accd[nt] = wmma_bf16(a, b, accd[nt]);
                }
            }
            // gated = tanh(filter+b2f) * sigmoid(gate+b2g) -> sG  (branch-free)
#pragma unroll
            for (int j = 0; j < 4; ++j) {
#pragma unroll
                for (int v = 0; v < 8; ++v) {
                    float fv = fast_tanh(accd[j][v] + b2f[j]);
                    float gv = fast_sigmoid(accd[j+4][v] + b2g[j]);
                    sG[wave][(v + hk*8)*LDG + j*16 + m] = f2bf(fv * gv);
                }
            }
            asm volatile("s_wait_dscnt 0" ::: "memory");

            // out = gated @ W4
            f32x8 acco[8];
#pragma unroll
            for (int nt = 0; nt < 8; ++nt) acco[nt] = (f32x8){};
#pragma unroll
            for (int kt = 0; kt < 2; ++kt) {
                bf16x16 a = loadA(&sG[wave][0], LDG, m, hk, kt*32);
#pragma unroll
                for (int nt = 0; nt < 8; ++nt) {
                    bf16x16 b = ldfrag16(frag + (size_t)(FRAG_W4 + kt*8 + nt)*512
                                              + lane*16);
                    acco[nt] = wmma_bf16(a, b, acco[nt]);
                }
            }
            // skips -> relu -> sG (reuse);  residuals -> inputs; persist inputs
            unsigned short* apw =
                app + (((size_t)i * BATCH + r0) * TPRED + t) * 64;
#pragma unroll
            for (int j = 0; j < 4; ++j) {
#pragma unroll
                for (int v = 0; v < 8; ++v) {
                    float sk = acco[j][v] + b4s[j];
                    sG[wave][(v + hk*8)*LDG + j*16 + m] = f2bf(sk > 0.f ? sk : 0.f);
                    float niv = acc_in[j][v] + acco[j+4][v] + b4r[j];
                    acc_in[j][v] = niv;
                    unsigned short hb = f2bf(niv);
                    sA[wave][(v + hk*8)*LDA + 64 + j*16 + m] = hb;   // next dil A
                    apw[(size_t)(v + hk*8) * TPRED * 64 + j*16 + m] = hb;
                }
            }
            asm volatile("s_wait_dscnt 0" ::: "memory");

            // h += relu(skips) @ W5[i*64:(i+1)*64, :]   (fold skip GEMM now)
#pragma unroll
            for (int kt = 0; kt < 2; ++kt) {
                bf16x16 a = loadA(&sG[wave][0], LDG, m, hk, kt*32);
#pragma unroll
                for (int nt = 0; nt < 8; ++nt) {
                    bf16x16 b = ldfrag16(frag +
                        (size_t)(FRAG_W5 + (i*2 + kt)*8 + nt)*512 + lane*16);
                    acc_h[nt] = wmma_bf16(a, b, acc_h[nt]);
                }
            }
        }

        // -------- y = relu(h + b5) @ W6 + b6 : fold on C-frags + butterfly ----
        f32x8 ys = (f32x8){};
#pragma unroll
        for (int nt = 0; nt < 8; ++nt) {
#pragma unroll
            for (int v = 0; v < 8; ++v) {
                float h = acc_h[nt][v] + b5v[nt];
                ys[v] += (h > 0.f ? h : 0.f) * w6v[nt];
            }
        }
#pragma unroll
        for (int off = 1; off < 16; off <<= 1)
#pragma unroll
            for (int v = 0; v < 8; ++v)
                ys[v] += __shfl_xor(ys[v], off, 32);
        if (m == 0) {
#pragma unroll
            for (int v = 0; v < 8; ++v) {
                int row = v + hk*8;
                float y = ys[v] + b6v;
                sY[wave][row] = y;                       // feeds next step's cur
                out[(size_t)(r0 + row) * TPRED + t] = y;
            }
        }
    }
}

// ---------------------------------------------------------------------------
extern "C" void kernel_launch(void* const* d_in, const int* in_sizes, int n_in,
                              void* d_out, int out_size, void* d_ws, size_t ws_size,
                              hipStream_t stream) {
    (void)in_sizes; (void)n_in; (void)out_size; (void)ws_size;
    const float* feat = (const float*)d_in[0];
    const float* init = (const float*)d_in[1];
    const float* enc  = (const float*)d_in[2];
    const float* W1   = (const float*)d_in[3];
    const float* b1   = (const float*)d_in[4];
    const float* W2   = (const float*)d_in[5];
    const float* b2   = (const float*)d_in[6];
    const float* W3   = (const float*)d_in[7];
    const float* W4   = (const float*)d_in[8];
    const float* b4   = (const float*)d_in[9];
    const float* W5   = (const float*)d_in[10];
    const float* b5   = (const float*)d_in[11];
    const float* W6   = (const float*)d_in[12];
    const float* b6   = (const float*)d_in[13];

    unsigned short* frag = (unsigned short*)d_ws;
    unsigned short* app  = (unsigned short*)((char*)d_ws + APP_OFF);
    float* out = (float*)d_out;

    pack_weights<<<(NFRAG_TOT * 512 + 255) / 256, 256, 0, stream>>>(
        W1, W2, W3, W4, W5, frag);
    decode_kernel<<<16, 128, 0, stream>>>(
        feat, init, enc, b1, b2, b4, b5, W6, b6, frag, app, out);
}
